// OptimizedTokenRoutedMLP_90383291777475
// MI455X (gfx1250) — matrix operand order repaired and checked by reading, not compiled, not run on hardware
//
#include <hip/hip_runtime.h>

typedef __attribute__((ext_vector_type(4)))  _Float16 v4h;
typedef __attribute__((ext_vector_type(8)))  _Float16 v8h;
typedef __attribute__((ext_vector_type(16))) _Float16 v16h;
typedef __attribute__((ext_vector_type(8)))  float    v8f;

#define HID    2048
#define EINT   1024
#define NEXP   8
#define NTOK   16384
#define NVOCAB 100000
#define LDA    40   // padded K-stride in halves (80 B: 16B-aligned, bank-conflict-free)

// ---------------- routing ----------------

__global__ void moe_init(int* meta) {
    if (threadIdx.x < 32) meta[threadIdx.x] = 0;
}

__global__ void moe_count(const int* __restrict__ tok, const int* __restrict__ t2e,
                          int* __restrict__ cnt) {
    int i = blockIdx.x * 256 + threadIdx.x;
    if (i >= NTOK) return;
    int t = tok[i];
    t = t < 0 ? 0 : (t > NVOCAB - 1 ? NVOCAB - 1 : t);
    atomicAdd(&cnt[t2e[t]], 1);
}

__global__ void moe_scan(int* meta) {
    // meta[0..7]=counts, meta[8..16]=offsets, meta[17..24]=cursors
    int acc = 0;
    for (int e = 0; e < NEXP; ++e) {
        meta[8 + e]  = acc;
        meta[17 + e] = acc;
        acc += meta[e];
    }
    meta[16] = acc;
}

__global__ void moe_scatter(const int* __restrict__ tok, const int* __restrict__ t2e,
                            int* __restrict__ cur, int* __restrict__ bucket) {
    int i = blockIdx.x * 256 + threadIdx.x;
    if (i >= NTOK) return;
    int t = tok[i];
    t = t < 0 ? 0 : (t > NVOCAB - 1 ? NVOCAB - 1 : t);
    int e = t2e[t];
    int pos = atomicAdd(&cur[e], 1);
    bucket[pos] = i;
}

// ---------------- GEMM1: gate+up fused, silu(g)*u -> f16 act ----------------
// Block tile M=128 x N=128, K-step 32. 8 waves: 4(M) x 2(N), wave tile 32x64.
// Register double-buffered staging; gather rows clamped (no exec divergence):
// an over-count A row only affects C rows that the epilogue masks anyway.

__global__ __launch_bounds__(256)
void moe_gemm1(const float* __restrict__ x, const float* __restrict__ Wg,
               const float* __restrict__ Wu, const int* __restrict__ bucket,
               const int* __restrict__ cnt, const int* __restrict__ offs,
               _Float16* __restrict__ act) {
    const int e     = blockIdx.z;
    const int count = cnt[e];
    const int m0    = blockIdx.y * 128;
    if (m0 >= count) return;
    const int seg   = offs[e];
    const int nbase = blockIdx.x * 128;

    __shared__ _Float16 ldsA[128 * LDA];
    __shared__ _Float16 ldsBg[128 * LDA];
    __shared__ _Float16 ldsBu[128 * LDA];

    const int t     = threadIdx.x;
    const int lane  = t & 31;
    const int wid   = t >> 5;
    const int wm    = wid >> 1;           // 0..3 -> 32-row sub-tile
    const int wn    = wid & 1;            // 0..1 -> 64-col sub-tile
    const int l16   = lane & 15;
    const int lhalf = (lane < 16) ? 0 : 1;

    // A staging: pass p handles row (t>>3)+32p, 4 consecutive K at (t&7)*4
    const int akq = (t & 7) * 4;
    size_t arow[4];
    #pragma unroll
    for (int p = 0; p < 4; ++p) {
        int r = m0 + (t >> 3) + 32 * p;
        r = r < count ? r : count - 1;    // clamp: keeps loads valid, no divergence
        arow[p] = (size_t)bucket[seg + r];
    }
    // B staging: pass p handles k = wid + 8p, 4 consecutive N at lane*4
    const int bnq = lane * 4;

    const float* wg = Wg + (size_t)e * HID * EINT;
    const float* wu = Wu + (size_t)e * HID * EINT;

    v8f accg[2][4], accu[2][4];
    #pragma unroll
    for (int am = 0; am < 2; ++am)
        #pragma unroll
        for (int f = 0; f < 4; ++f)
            #pragma unroll
            for (int i = 0; i < 8; ++i) { accg[am][f][i] = 0.f; accu[am][f][i] = 0.f; }

    // register staging buffers (double buffer vs. LDS)
    float4 ra[4], rg[4], ru[4];
    #pragma unroll
    for (int p = 0; p < 4; ++p) {
        ra[p] = *(const float4*)(x + arow[p] * HID + akq);
        const size_t gi = (size_t)(wid + 8 * p) * EINT + nbase + bnq;
        rg[p] = *(const float4*)(wg + gi);
        ru[p] = *(const float4*)(wu + gi);
    }

    for (int kb = 0; kb < HID; kb += 32) {
        __syncthreads();   // previous compute done reading LDS
        // ---- commit staged registers to LDS (fp32 -> f16) ----
        #pragma unroll
        for (int p = 0; p < 4; ++p) {
            v4h h; h[0] = (_Float16)ra[p].x; h[1] = (_Float16)ra[p].y;
                   h[2] = (_Float16)ra[p].z; h[3] = (_Float16)ra[p].w;
            *(v4h*)(&ldsA[((t >> 3) + 32 * p) * LDA + akq]) = h;
        }
        #pragma unroll
        for (int p = 0; p < 4; ++p) {
            const int k = wid + 8 * p;
            ldsBg[(bnq + 0) * LDA + k] = (_Float16)rg[p].x;
            ldsBg[(bnq + 1) * LDA + k] = (_Float16)rg[p].y;
            ldsBg[(bnq + 2) * LDA + k] = (_Float16)rg[p].z;
            ldsBg[(bnq + 3) * LDA + k] = (_Float16)rg[p].w;
            ldsBu[(bnq + 0) * LDA + k] = (_Float16)ru[p].x;
            ldsBu[(bnq + 1) * LDA + k] = (_Float16)ru[p].y;
            ldsBu[(bnq + 2) * LDA + k] = (_Float16)ru[p].z;
            ldsBu[(bnq + 3) * LDA + k] = (_Float16)ru[p].w;
        }
        __syncthreads();   // staged tile visible

        // ---- issue next tile's global loads (overlap with WMMA below) ----
        if (kb + 32 < HID) {
            const int kn = kb + 32;
            #pragma unroll
            for (int p = 0; p < 4; ++p) {
                ra[p] = *(const float4*)(x + arow[p] * HID + kn + akq);
                const size_t gi = (size_t)(kn + wid + 8 * p) * EINT + nbase + bnq;
                rg[p] = *(const float4*)(wg + gi);
                ru[p] = *(const float4*)(wu + gi);
            }
        }

        // ---- A fragments (2x 16x32): lane<16: K{0..7,16..23}; lane>=16: K{8..15,24..31} ----
        v16h a[2];
        #pragma unroll
        for (int am = 0; am < 2; ++am) {
            const _Float16* ab = &ldsA[(wm * 32 + am * 16 + l16) * LDA + lhalf * 8];
            v8h alo = *(const v8h*)(ab);
            v8h ahi = *(const v8h*)(ab + 16);
            #pragma unroll
            for (int i = 0; i < 8; ++i) { a[am][i] = alo[i]; a[am][i + 8] = ahi[i]; }
        }

        #pragma unroll
        for (int f = 0; f < 4; ++f) {
            const int col = (wn * 64 + f * 16 + l16) * LDA + lhalf * 8;
            v8h blo = *(const v8h*)(&ldsBg[col]);
            v8h bhi = *(const v8h*)(&ldsBg[col + 16]);
            v16h b;
            #pragma unroll
            for (int i = 0; i < 8; ++i) { b[i] = blo[i]; b[i + 8] = bhi[i]; }
            #pragma unroll
            for (int am = 0; am < 2; ++am)
                accg[am][f] = __builtin_amdgcn_wmma_f32_16x16x32_f16(false, a[am], false, b,
                                                                    (short)0, accg[am][f], false, false);
            blo = *(const v8h*)(&ldsBu[col]);
            bhi = *(const v8h*)(&ldsBu[col + 16]);
            #pragma unroll
            for (int i = 0; i < 8; ++i) { b[i] = blo[i]; b[i + 8] = bhi[i]; }
            #pragma unroll
            for (int am = 0; am < 2; ++am)
                accu[am][f] = __builtin_amdgcn_wmma_f32_16x16x32_f16(false, a[am], false, b,
                                                                    (short)0, accu[am][f], false, false);
        }
    }

    // ---- epilogue: silu(g)*u -> f16 act at gathered positions ----
    #pragma unroll
    for (int am = 0; am < 2; ++am)
        #pragma unroll
        for (int f = 0; f < 4; ++f) {
            const int col = nbase + wn * 64 + f * 16 + l16;
            #pragma unroll
            for (int r = 0; r < 8; ++r) {
                const int gr = m0 + wm * 32 + am * 16 + r + lhalf * 8;
                if (gr < count) {
                    const float g = accg[am][f][r];
                    const float u = accu[am][f][r];
                    const float h = (g / (1.f + __expf(-g))) * u;
                    act[(size_t)(seg + gr) * EINT + col] = (_Float16)h;
                }
            }
        }
}

// ---------------- GEMM2: act @ down -> scatter fp32 out ----------------
// Block tile M=128 x N=128, K-step 32. 8 waves: 4(M) x 2(N).

__global__ __launch_bounds__(256)
void moe_gemm2(const _Float16* __restrict__ act, const float* __restrict__ Wd,
               const int* __restrict__ bucket, const int* __restrict__ cnt,
               const int* __restrict__ offs, float* __restrict__ out) {
    const int e     = blockIdx.z;
    const int count = cnt[e];
    const int m0    = blockIdx.y * 128;
    if (m0 >= count) return;
    const int seg   = offs[e];
    const int nbase = blockIdx.x * 128;

    __shared__ _Float16 ldsA[128 * LDA];
    __shared__ _Float16 ldsB[128 * LDA];

    const int t     = threadIdx.x;
    const int lane  = t & 31;
    const int wid   = t >> 5;
    const int wm    = wid >> 1;
    const int wn    = wid & 1;
    const int l16   = lane & 15;
    const int lhalf = (lane < 16) ? 0 : 1;

    // A staging (f16): pass p handles row (t>>2)+64p, 8 consecutive halves at (t&3)*8
    const int ahq = (t & 3) * 8;
    size_t aro[2];
    #pragma unroll
    for (int p = 0; p < 2; ++p) {
        int r = m0 + (t >> 2) + 64 * p;
        r = r < count ? r : count - 1;    // clamp: masked at epilogue
        aro[p] = (size_t)(seg + r) * EINT;
    }
    // B staging: pass p handles k = wid + 8p, 4 consecutive N at lane*4
    const int bnq = lane * 4;

    const float* wd = Wd + (size_t)e * EINT * HID;

    v8f acc[2][4];
    #pragma unroll
    for (int am = 0; am < 2; ++am)
        #pragma unroll
        for (int f = 0; f < 4; ++f)
            #pragma unroll
            for (int i = 0; i < 8; ++i) acc[am][f][i] = 0.f;

    // register staging buffers
    v8h    ra[2];
    float4 rb[4];
    #pragma unroll
    for (int p = 0; p < 2; ++p) ra[p] = *(const v8h*)(act + aro[p] + ahq);
    #pragma unroll
    for (int p = 0; p < 4; ++p)
        rb[p] = *(const float4*)(wd + (size_t)(wid + 8 * p) * HID + nbase + bnq);

    for (int kb = 0; kb < EINT; kb += 32) {
        __syncthreads();
        // ---- commit staged registers to LDS ----
        #pragma unroll
        for (int p = 0; p < 2; ++p)
            *(v8h*)(&ldsA[((t >> 2) + 64 * p) * LDA + ahq]) = ra[p];
        #pragma unroll
        for (int p = 0; p < 4; ++p) {
            const int k = wid + 8 * p;
            ldsB[(bnq + 0) * LDA + k] = (_Float16)rb[p].x;
            ldsB[(bnq + 1) * LDA + k] = (_Float16)rb[p].y;
            ldsB[(bnq + 2) * LDA + k] = (_Float16)rb[p].z;
            ldsB[(bnq + 3) * LDA + k] = (_Float16)rb[p].w;
        }
        __syncthreads();

        // ---- issue next tile's global loads ----
        if (kb + 32 < EINT) {
            const int kn = kb + 32;
            #pragma unroll
            for (int p = 0; p < 2; ++p) ra[p] = *(const v8h*)(act + aro[p] + kn + ahq);
            #pragma unroll
            for (int p = 0; p < 4; ++p)
                rb[p] = *(const float4*)(wd + (size_t)(kn + wid + 8 * p) * HID + nbase + bnq);
        }

        v16h a[2];
        #pragma unroll
        for (int am = 0; am < 2; ++am) {
            const _Float16* ab = &ldsA[(wm * 32 + am * 16 + l16) * LDA + lhalf * 8];
            v8h alo = *(const v8h*)(ab);
            v8h ahi = *(const v8h*)(ab + 16);
            #pragma unroll
            for (int i = 0; i < 8; ++i) { a[am][i] = alo[i]; a[am][i + 8] = ahi[i]; }
        }

        #pragma unroll
        for (int f = 0; f < 4; ++f) {
            const int col = (wn * 64 + f * 16 + l16) * LDA + lhalf * 8;
            v8h blo = *(const v8h*)(&ldsB[col]);
            v8h bhi = *(const v8h*)(&ldsB[col + 16]);
            v16h b;
            #pragma unroll
            for (int i = 0; i < 8; ++i) { b[i] = blo[i]; b[i + 8] = bhi[i]; }
            #pragma unroll
            for (int am = 0; am < 2; ++am)
                acc[am][f] = __builtin_amdgcn_wmma_f32_16x16x32_f16(false, a[am], false, b,
                                                                   (short)0, acc[am][f], false, false);
        }
    }

    // ---- epilogue: scatter fp32 rows back to token order ----
    #pragma unroll
    for (int am = 0; am < 2; ++am)
        #pragma unroll
        for (int f = 0; f < 4; ++f) {
            const int col = nbase + wn * 64 + f * 16 + l16;
            #pragma unroll
            for (int r = 0; r < 8; ++r) {
                const int gr = m0 + wm * 32 + am * 16 + r + lhalf * 8;
                if (gr < count) {
                    const int tokrow = bucket[seg + gr];
                    out[(size_t)tokrow * HID + col] = acc[am][f][r];
                }
            }
        }
}

// ---------------- launch ----------------

extern "C" void kernel_launch(void* const* d_in, const int* in_sizes, int n_in,
                              void* d_out, int out_size, void* d_ws, size_t ws_size,
                              hipStream_t stream) {
    (void)in_sizes; (void)n_in; (void)out_size; (void)ws_size;
    const float* x   = (const float*)d_in[0];   // [4,4096,2048]
    const float* wg  = (const float*)d_in[1];   // [8,2048,1024]
    const float* wu  = (const float*)d_in[2];   // [8,2048,1024]
    const float* wd  = (const float*)d_in[3];   // [8,1024,2048]
    const int*   tok = (const int*)d_in[4];     // [4,4096]
    const int*   t2e = (const int*)d_in[5];     // [100000]
    float*       out = (float*)d_out;

    int*      meta   = (int*)d_ws;                               // 32 ints: cnt[8], offs[9], cur[8]
    int*      bucket = (int*)((char*)d_ws + 256);                // NTOK ints
    _Float16* act    = (_Float16*)((char*)d_ws + 256 + NTOK * sizeof(int)); // NTOK x EINT f16 (~33.5 MB)

    moe_init<<<1, 32, 0, stream>>>(meta);
    moe_count<<<NTOK / 256, 256, 0, stream>>>(tok, t2e, meta);
    moe_scan<<<1, 1, 0, stream>>>(meta);
    moe_scatter<<<NTOK / 256, 256, 0, stream>>>(tok, t2e, meta + 17, bucket);
    moe_gemm1<<<dim3(EINT / 128, NTOK / 128, NEXP), 256, 0, stream>>>(x, wg, wu, bucket, meta, meta + 8, act);
    moe_gemm2<<<dim3(HID / 128, NTOK / 128, NEXP), 256, 0, stream>>>(act, wd, bucket, meta, meta + 8, out);
}
